// AtomBlock_14791867367765
// MI455X (gfx1250) — matrix operand order repaired and checked by reading, not compiled, not run on hardware
//
#include <hip/hip_runtime.h>
#include <math.h>

// ---------------------------------------------------------------------------
// AtomBlock (AF3-style atom attention) for gfx1250.
// All dense projections run through v_wmma_f32_16x16x32_f16 (wave32 WMMA).
// B is staged TRANSPOSED in LDS so fragments load as wide ds_load_b128 ops;
// all fragments are preloaded per K-step so the 4 WMMAs issue back-to-back.
// Attention exploits the +/-16 band: 33 keys per row, one wave per (row,head).
// ---------------------------------------------------------------------------

typedef __attribute__((ext_vector_type(16))) _Float16 v16h;
typedef __attribute__((ext_vector_type(8)))  _Float16 v8h;
typedef __attribute__((ext_vector_type(8)))  float    v8f;
typedef __attribute__((ext_vector_type(4)))  float    v4f;

#define NA   4096
#define DA   128
#define DM   512
#define HEADS 4
#define DH   32
#define DF   512
#define PNUM 32768
#define WIN  16
#define BANDW 33

#define TM 64
#define TN 64
#define TK 32
#define LDA 40   // padded LDS stride (halves): 80B -> conflict-free & 16B aligned
#define LDB 40

// ---------------------------------------------------------------------------
// Generic WMMA GEMM: C[M,N] = op(A)[M,K] @ W[K,N] (+ bias[col] + bias2[col]).
// op(A) row i = A[gidx[i]] when gidx != nullptr (token gather), else A[i].
// 128 threads = 4 waves; 64x64 C tile; K stepped by 32 through LDS with
// fp32 -> fp16 conversion; fp32 WMMA accumulators.
// ---------------------------------------------------------------------------
__global__ __launch_bounds__(128)
void gemm_wmma(const float* __restrict__ A, const float* __restrict__ W,
               const float* __restrict__ bias, const float* __restrict__ bias2,
               float* __restrict__ C, int M, int K, int N,
               const int* __restrict__ gidx)
{
    __shared__ _Float16 lA[TM * LDA];   // [row][k]
    __shared__ _Float16 lB[TN * LDB];   // [n][k]  (TRANSPOSED)

    const int tid  = threadIdx.x;
    const int lane = tid & 31;
    const int wv   = tid >> 5;
    const int m0   = blockIdx.x * TM;
    const int n0   = blockIdx.y * TN;

    const int mrow = lane & 15;    // M (A) or N (B/C) index within 16-tile
    const int hi   = lane >> 4;    // lane-half selector

    v8f acc[4] = {};

    // A staging: thread -> (row 0..63, 16-float chunk)
    const int sArow = tid >> 1;
    const int sAcol = (tid & 1) * 16;
    int aRowG = m0 + sArow;
    if (gidx) aRowG = gidx[aRowG];
    const float* Arow = A + (long)aRowG * K;

    // B staging: thread -> (column n 0..63, 16 consecutive k)
    const int sBn = tid & 63;
    const int sBk = (tid >> 6) * 16;

    for (int k0 = 0; k0 < K; k0 += TK) {
        // ---- stage A (row-major): 4x v4f global -> 2x v8h LDS
        {
            const float* src = Arow + k0 + sAcol;
            v4f f0 = *(const v4f*)(src + 0);
            v4f f1 = *(const v4f*)(src + 4);
            v4f f2 = *(const v4f*)(src + 8);
            v4f f3 = *(const v4f*)(src + 12);
            v8h h0, h1;
            #pragma unroll
            for (int e = 0; e < 4; ++e) {
                h0[e]     = (_Float16)f0[e];
                h0[4 + e] = (_Float16)f1[e];
                h1[e]     = (_Float16)f2[e];
                h1[4 + e] = (_Float16)f3[e];
            }
            _Float16* dst = &lA[sArow * LDA + sAcol];
            *(v8h*)(dst)     = h0;
            *(v8h*)(dst + 8) = h1;
        }
        // ---- stage B transposed: 16 coalesced row reads -> 2x v8h LDS
        {
            const float* src = W + (long)(k0 + sBk) * N + n0 + sBn;
            v8h h0, h1;
            #pragma unroll
            for (int kk = 0; kk < 8; ++kk) {
                h0[kk] = (_Float16)src[(long)kk * N];
                h1[kk] = (_Float16)src[(long)(kk + 8) * N];
            }
            _Float16* dst = &lB[sBn * LDB + sBk];
            *(v8h*)(dst)     = h0;
            *(v8h*)(dst + 8) = h1;
        }
        __syncthreads();

        // ---- A fragment (16x32 f16): lane m holds K = hi*16+[0..15]
        const _Float16* aptr = &lA[(wv * 16 + mrow) * LDA];
        v8h a0 = *(const v8h*)(aptr + hi * 16);
        v8h a1 = *(const v8h*)(aptr + hi * 16 + 8);
        v16h af;
        #pragma unroll
        for (int e = 0; e < 8; ++e) { af[e] = a0[e]; af[8 + e] = a1[e]; }

        // ---- preload ALL B fragments (32x16 f16): lane n holds K = hi*16+[0..15]
        v16h bf[4];
        #pragma unroll
        for (int t = 0; t < 4; ++t) {
            const _Float16* bptr = &lB[(t * 16 + mrow) * LDB + hi * 16];
            v8h b0 = *(const v8h*)(bptr);
            v8h b1 = *(const v8h*)(bptr + 8);
            #pragma unroll
            for (int e = 0; e < 8; ++e) { bf[t][e] = b0[e]; bf[t][8 + e] = b1[e]; }
        }

        // ---- dense WMMA chain (back-to-back matrix issue)
        #pragma unroll
        for (int t = 0; t < 4; ++t) {
            acc[t] = __builtin_amdgcn_wmma_f32_16x16x32_f16(
                false, af, false, bf[t], (short)0, acc[t], false, false);
        }
        __syncthreads();
    }

    // epilogue: C layout — VGPR r holds M = hi*8 + r, N = lane&15
    #pragma unroll
    for (int t = 0; t < 4; ++t) {
        const int col = n0 + t * 16 + mrow;
        float badd = 0.f;
        if (bias)  badd += bias[col];
        if (bias2) badd += bias2[col];
        #pragma unroll
        for (int r = 0; r < 8; ++r) {
            const int row = m0 + wv * 16 + hi * 8 + r;
            C[(long)row * N + col] = acc[t][r] + badd;
        }
    }
}

// ---------------------------------------------------------------------------
// adaLN + LayerNorm: out = (1 + ad[:, :128]) * LN(x) + ad[:, 128:256]
// one wave per row (lane handles 4 of 128 channels), butterfly reductions.
// ---------------------------------------------------------------------------
__global__ __launch_bounds__(256)
void adaln_ln(const float* __restrict__ x, const float* __restrict__ ad,
              const float* __restrict__ lng, const float* __restrict__ lnb,
              float* __restrict__ out, int useAffine)
{
    const int lane = threadIdx.x & 31;
    const int w    = threadIdx.x >> 5;
    const long row = (long)blockIdx.x * 8 + w;
    const float* xr = x + row * DA;

    float v[4], s = 0.f, sq = 0.f;
    #pragma unroll
    for (int e = 0; e < 4; ++e) {
        v[e] = xr[lane + 32 * e];
        s += v[e]; sq += v[e] * v[e];
    }
    #pragma unroll
    for (int off = 16; off > 0; off >>= 1) {
        s  += __shfl_xor(s,  off, 32);
        sq += __shfl_xor(sq, off, 32);
    }
    const float mean = s * (1.f / DA);
    const float var  = sq * (1.f / DA) - mean * mean;
    const float rstd = rsqrtf(var + 1e-5f);

    const float* adr = ad + row * (2 * DA);
    #pragma unroll
    for (int e = 0; e < 4; ++e) {
        const int c = lane + 32 * e;
        float xn = (v[e] - mean) * rstd;
        if (useAffine) xn = xn * lng[c] + lnb[c];
        out[row * DA + c] = (1.f + adr[c]) * xn + adr[DA + c];
    }
}

// ---------------------------------------------------------------------------
// Pair bias scatter into the attention band (only |i-j|<=16 entries matter).
// bias[p,h] = p_lm[p,:16] @ pair_w[:,h] + pair_b[h]
// ---------------------------------------------------------------------------
__global__ void pair_scatter(const float* __restrict__ plm, const float* __restrict__ pw,
                             const float* __restrict__ pb, const int* __restrict__ pidx,
                             float* __restrict__ band)
{
    const int p = blockIdx.x * blockDim.x + threadIdx.x;
    if (p >= PNUM) return;
    const int i = pidx[2 * p], j = pidx[2 * p + 1];
    const int d = j - i;
    if (d < -WIN || d > WIN) return;
    const float* f = plm + (long)p * 16;
    #pragma unroll
    for (int h = 0; h < HEADS; ++h) {
        float s = pb[h];
        #pragma unroll
        for (int k = 0; k < 16; ++k) s += f[k] * pw[k * HEADS + h];
        band[((long)i * HEADS + h) * BANDW + d + WIN] = s;   // .set semantics
    }
}

// ---------------------------------------------------------------------------
// Banded softmax attention: one wave per (row i, head h), lane = head-dim.
// scores over j in [i-16, i+16]; softmax over <=33 entries via per-wave LDS.
// ---------------------------------------------------------------------------
__global__ __launch_bounds__(256)
void attn_band(const float* __restrict__ Qm, const float* __restrict__ Km,
               const float* __restrict__ Vm, const float* __restrict__ band,
               float* __restrict__ out)
{
    __shared__ float sw[8][40];
    const int lane = threadIdx.x & 31;
    const int w    = threadIdx.x >> 5;
    const int gw   = blockIdx.x * 8 + w;     // 0 .. NA*HEADS-1
    const int i    = gw >> 2;
    const int h    = gw & 3;
    const float scale = 0.17677669529663687f;  // 1/sqrt(32)

    const float qv = Qm[(long)i * DA + h * DH + lane];
    int jlo = i - WIN; if (jlo < 0) jlo = 0;
    int jhi = i + WIN; if (jhi > NA - 1) jhi = NA - 1;
    const int nw = jhi - jlo + 1;

    for (int j = jlo; j <= jhi; ++j) {
        float t = qv * Km[(long)j * DA + h * DH + lane];
        #pragma unroll
        for (int off = 16; off > 0; off >>= 1) t += __shfl_xor(t, off, 32);
        if (lane == 0)
            sw[w][j - jlo] = t * scale + band[((long)i * HEADS + h) * BANDW + (j - i + WIN)];
    }
    __syncthreads();

    float m = -3.4e38f;
    for (int x = 0; x < nw; ++x) m = fmaxf(m, sw[w][x]);
    float Z = 0.f;
    for (int x = 0; x < nw; ++x) Z += expf(sw[w][x] - m);
    float acc = 0.f;
    for (int x = 0; x < nw; ++x)
        acc += expf(sw[w][x] - m) * Vm[(long)(jlo + x) * DA + h * DH + lane];
    out[(long)i * DA + h * DH + lane] = acc / Z;
}

// ---------------------------------------------------------------------------
// Elementwise epilogues
// ---------------------------------------------------------------------------
__device__ __forceinline__ float sigm(float x) { return 1.f / (1.f + expf(-x)); }

__global__ void zero_f32(float* __restrict__ p, int n) {
    const int i = blockIdx.x * blockDim.x + threadIdx.x;
    if (i < n) p[i] = 0.f;
}

// q1 = (q + sigmoid(G) * proj) * (1 + sigmoid(g1v))
__global__ void ew_q1(const float* __restrict__ q, const float* __restrict__ G,
                      const float* __restrict__ proj, const float* __restrict__ g1v,
                      float* __restrict__ q1, int n) {
    const int i = blockIdx.x * blockDim.x + threadIdx.x;
    if (i < n) q1[i] = (q[i] + sigm(G[i]) * proj[i]) * (1.f + sigm(g1v[i]));
}

// h = silu(s1) * s3   (in-place into s1 is safe)
__global__ void ew_silumul(const float* __restrict__ s1, const float* __restrict__ s3,
                           float* __restrict__ o, int n) {
    const int i = blockIdx.x * blockDim.x + threadIdx.x;
    if (i < n) { const float a = s1[i]; o[i] = a * sigm(a) * s3[i]; }
}

// out = q1 + sigmoid(g2v) * swi
__global__ void ew_final(const float* __restrict__ q1, const float* __restrict__ g2v,
                         const float* __restrict__ swi, float* __restrict__ out, int n) {
    const int i = blockIdx.x * blockDim.x + threadIdx.x;
    if (i < n) out[i] = q1[i] + sigm(g2v[i]) * swi[i];
}

// ---------------------------------------------------------------------------
// Workspace layout (floats)
// ---------------------------------------------------------------------------
#define OFF_COND 0L                         // 4096*128
#define OFF_AD   524288L                    // 4096*256 (shared adaln1/adaln2)
#define OFF_QN   1572864L                   // 4096*128 (shared qn / qn2)
#define OFF_Q    2097152L
#define OFF_K    2621440L
#define OFF_V    3145728L
#define OFF_G    3670016L
#define OFF_BAND 4194304L                   // 4096*4*33 = 540672
#define OFF_ATT  4734976L
#define OFF_PROJ 5259264L
#define OFF_G1V  5783552L
#define OFF_Q1   6307840L
#define OFF_S1   6832128L                   // 4096*512
#define OFF_S3   8929280L                   // 4096*512
#define OFF_SWI  OFF_Q                      // Q free after attention
#define OFF_G2V  OFF_K                      // K free after attention

extern "C" void kernel_launch(void* const* d_in, const int* in_sizes, int n_in,
                              void* d_out, int out_size, void* d_ws, size_t ws_size,
                              hipStream_t stream)
{
    (void)in_sizes; (void)n_in; (void)out_size; (void)ws_size;
    const float* q           = (const float*)d_in[0];
    const float* c_atom      = (const float*)d_in[1];
    const float* h_cond      = (const float*)d_in[2];
    const float* p_lm        = (const float*)d_in[3];
    const float* t_emb       = (const float*)d_in[4];
    const float* cond_proj_w = (const float*)d_in[5];
    const float* cond_proj_b = (const float*)d_in[6];
    const float* adaln1_w    = (const float*)d_in[7];
    const float* adaln1_b    = (const float*)d_in[8];
    const float* adaln2_w    = (const float*)d_in[9];
    const float* adaln2_b    = (const float*)d_in[10];
    const float* ln_g        = (const float*)d_in[11];
    const float* ln_b        = (const float*)d_in[12];
    const float* wq          = (const float*)d_in[13];
    const float* wk          = (const float*)d_in[14];
    const float* wvp         = (const float*)d_in[15];
    const float* wg          = (const float*)d_in[16];
    const float* wo          = (const float*)d_in[17];
    const float* pair_w      = (const float*)d_in[18];
    const float* pair_b      = (const float*)d_in[19];
    const float* gate1_w     = (const float*)d_in[20];
    const float* gate1_b     = (const float*)d_in[21];
    const float* gate2_w     = (const float*)d_in[22];
    const float* gate2_b     = (const float*)d_in[23];
    const float* sw1         = (const float*)d_in[24];
    const float* sw3         = (const float*)d_in[25];
    const float* sw2         = (const float*)d_in[26];
    const int*   p_lm_idx    = (const int*)d_in[27];
    const int*   token_idx   = (const int*)d_in[28];

    float* ws  = (float*)d_ws;
    float* out = (float*)d_out;

    const int nED = NA * DA;          // 524288
    const int nEF = NA * DF;          // 2097152
    const int nBand = NA * HEADS * BANDW;

    const dim3 blkG(128);
    const dim3 gN128(NA / TM, DA / TN);        // 64 x 2
    const dim3 gN256(NA / TM, 256 / TN);       // 64 x 4
    const dim3 gN512(NA / TM, DF / TN);        // 64 x 8

    // pair bias band
    zero_f32<<<(nBand + 255) / 256, 256, 0, stream>>>(ws + OFF_BAND, nBand);
    pair_scatter<<<(PNUM + 255) / 256, 256, 0, stream>>>(p_lm, pair_w, pair_b, p_lm_idx,
                                                         ws + OFF_BAND);

    // cond = t_emb + h_cond[token_idx] @ cond_proj_w + cond_proj_b
    gemm_wmma<<<gN128, blkG, 0, stream>>>(h_cond, cond_proj_w, cond_proj_b, t_emb,
                                          ws + OFF_COND, NA, DM, DA, token_idx);
    // adaln1 = cond @ adaln1_w + adaln1_b
    gemm_wmma<<<gN256, blkG, 0, stream>>>(ws + OFF_COND, adaln1_w, adaln1_b, nullptr,
                                          ws + OFF_AD, NA, DA, 2 * DA, nullptr);
    // q_n = (1+g1)*LN(q;g,b) + b1
    adaln_ln<<<NA / 8, 256, 0, stream>>>(q, ws + OFF_AD, ln_g, ln_b, ws + OFF_QN, 1);

    // Q, K, V, G projections
    gemm_wmma<<<gN128, blkG, 0, stream>>>(ws + OFF_QN, wq,  nullptr, nullptr, ws + OFF_Q, NA, DA, DA, nullptr);
    gemm_wmma<<<gN128, blkG, 0, stream>>>(ws + OFF_QN, wk,  nullptr, nullptr, ws + OFF_K, NA, DA, DA, nullptr);
    gemm_wmma<<<gN128, blkG, 0, stream>>>(ws + OFF_QN, wvp, nullptr, nullptr, ws + OFF_V, NA, DA, DA, nullptr);
    gemm_wmma<<<gN128, blkG, 0, stream>>>(ws + OFF_QN, wg,  nullptr, nullptr, ws + OFF_G, NA, DA, DA, nullptr);

    // banded attention
    attn_band<<<NA * HEADS / 8, 256, 0, stream>>>(ws + OFF_Q, ws + OFF_K, ws + OFF_V,
                                                  ws + OFF_BAND, ws + OFF_ATT);

    // output proj + gates -> q1
    gemm_wmma<<<gN128, blkG, 0, stream>>>(ws + OFF_ATT, wo, nullptr, nullptr, ws + OFF_PROJ, NA, DA, DA, nullptr);
    gemm_wmma<<<gN128, blkG, 0, stream>>>(c_atom, gate1_w, gate1_b, nullptr, ws + OFF_G1V, NA, DA, DA, nullptr);
    ew_q1<<<(nED + 255) / 256, 256, 0, stream>>>(q, ws + OFF_G, ws + OFF_PROJ, ws + OFF_G1V,
                                                 ws + OFF_Q1, nED);

    // adaln2 + LN(q1) -> q_n2
    gemm_wmma<<<gN256, blkG, 0, stream>>>(ws + OFF_COND, adaln2_w, adaln2_b, nullptr,
                                          ws + OFF_AD, NA, DA, 2 * DA, nullptr);
    adaln_ln<<<NA / 8, 256, 0, stream>>>(ws + OFF_Q1, ws + OFF_AD, nullptr, nullptr, ws + OFF_QN, 0);

    // SwiGLU
    gemm_wmma<<<gN512, blkG, 0, stream>>>(ws + OFF_QN, sw1, nullptr, nullptr, ws + OFF_S1, NA, DA, DF, nullptr);
    gemm_wmma<<<gN512, blkG, 0, stream>>>(ws + OFF_QN, sw3, nullptr, nullptr, ws + OFF_S3, NA, DA, DF, nullptr);
    ew_silumul<<<(nEF + 255) / 256, 256, 0, stream>>>(ws + OFF_S1, ws + OFF_S3, ws + OFF_S1, nEF);
    gemm_wmma<<<gN128, blkG, 0, stream>>>(ws + OFF_S1, sw2, nullptr, nullptr, ws + OFF_SWI, NA, DF, DA, nullptr);

    // final gate
    gemm_wmma<<<gN128, blkG, 0, stream>>>(c_atom, gate2_w, gate2_b, nullptr, ws + OFF_G2V, NA, DA, DA, nullptr);
    ew_final<<<(nED + 255) / 256, 256, 0, stream>>>(ws + OFF_Q1, ws + OFF_G2V, ws + OFF_SWI, out, nED);
}
